// CGCNN_609885356796
// MI455X (gfx1250) — compile-verified, matrix operand order checked by reading, not run on hardware
//
#include <hip/hip_runtime.h>
#include <hip/hip_bf16.h>
#include <math.h>

// ---------------------------------------------------------------------------
// CGCNN forward for MI455X (gfx1250, wave32, WMMA).
//
// Roofline: dominant GEMM = 1.2M x 169 -> 128 per conv layer (x3). Atom
// features (25.6MB) are L2-resident, nbr_fea (197MB) streams from HBM.
// BatchNorm needs global column stats -> two-pass recompute (cheap f16 WMMA
// flops) instead of materializing the 614MB gated tensor. Total HBM ~1.4GB
// => ~60us at 23.3TB/s; WMMA f16 (f32 acc) keeps compute off critical path.
// ---------------------------------------------------------------------------

#define EPSV 1e-5f
constexpr int N_ATOMS = 100000;
constexpr int M_NBR   = 12;
constexpr int B_CRYS  = 2000;
constexpr int F_INF   = 92;
constexpr int F_DIM   = 64;
constexpr int NBR_F   = 41;
constexpr int D_DIM   = 128;
constexpr int K_DIM   = 2 * F_DIM + NBR_F;  // 169
constexpr int K_PAD   = 192;                // 6 chunks of 32
constexpr int G_ATOMS = 4;
constexpr int G_ROWS  = G_ATOMS * M_NBR;    // 48 rows per group (3 frags of 16)
constexpr int N_GROUPS = N_ATOMS / G_ATOMS; // 25000
constexpr int WPACK_PER_LAYER = 6 * 8 * 32 * 8; // uints = 12288
constexpr int GSTRIDE = 132;                // padded gated-row stride (floats)

typedef _Float16 v16h __attribute__((ext_vector_type(16)));
typedef float    v8f  __attribute__((ext_vector_type(8)));

union Frag { v16h v; unsigned int u[8]; };
union HPack { _Float16 h[2]; unsigned int u; };

__device__ __forceinline__ float softplusf(float x) {
    return (x > 20.f) ? x : log1pf(__expf(x));
}
__device__ __forceinline__ float sigmoidf(float x) {
    return 1.f / (1.f + __expf(-x));
}

// -------------------------------- utility ----------------------------------
__global__ void zero_f32(float* p, int n) {
    int i = blockIdx.x * blockDim.x + threadIdx.x;
    if (i < n) p[i] = 0.f;
}

// mean/invstd from sum & sumsq
__global__ void finalize_stats(const float* __restrict__ s, const float* __restrict__ q,
                               float cnt, int n, float* __restrict__ mean,
                               float* __restrict__ inv) {
    int i = blockIdx.x * blockDim.x + threadIdx.x;
    if (i < n) {
        float m = s[i] / cnt;
        float v = q[i] / cnt - m * m;
        mean[i] = m;
        inv[i]  = rsqrtf(v + EPSV);
    }
}

// ---------------------- pack conv weights into B fragments -----------------
// wpack layout per layer: [kchunk c][coltile t][lane][vgpr v] -> uint (2 f16)
// B 32x16 f16 layout: lane 0-15 col=lane K=0..7,16..23 (pairs per VGPR),
//                     lane 16-31 col=lane-15.. K=8..15,24..31.
__global__ void pack_weights(const float* __restrict__ conv_w, unsigned int* __restrict__ wpack) {
    int idx = blockIdx.x * blockDim.x + threadIdx.x;
    if (idx >= 3 * WPACK_PER_LAYER) return;
    int v = idx & 7;
    int r = idx >> 3;
    int lane = r & 31;  r >>= 5;
    int t = r & 7;      r >>= 3;
    int c = r % 6;
    int layer = r / 6;
    int halfsel = lane >> 4;
    int cit = lane & 15;
    int k0 = c * 32 + ((v < 4) ? 0 : 16) + halfsel * 8 + (v & 3) * 2;
    int col = t * 16 + cit;
    const float* w = conv_w + (size_t)layer * K_DIM * D_DIM;
    HPack pk;
    pk.h[0] = (k0     < K_DIM) ? (_Float16)w[(size_t)k0 * D_DIM + col]       : (_Float16)0.f;
    pk.h[1] = (k0 + 1 < K_DIM) ? (_Float16)w[(size_t)(k0 + 1) * D_DIM + col] : (_Float16)0.f;
    wpack[idx] = pk.u;
}

// ------------------------------ embed: x @ w1 + b1 -------------------------
__global__ void embed_kernel(const float* __restrict__ x, const float* __restrict__ w1,
                             const float* __restrict__ b1, float* __restrict__ atomOut) {
    int idx = blockIdx.x * blockDim.x + threadIdx.x;
    if (idx >= N_ATOMS * F_DIM) return;
    int n = idx >> 6, f = idx & 63;
    float a = b1[f];
    const float* xr = x + (size_t)n * F_INF;
#pragma unroll 4
    for (int k = 0; k < F_INF; ++k) a += xr[k] * w1[k * F_DIM + f];
    atomOut[idx] = a;
}

// --------------------- shared A-fragment loader (LDS -> regs) --------------
// 16-bit A 16x32 layout: lane row = lane&15; lanes>=16 take K-halves +8.
__device__ __forceinline__ void load_afrag(const _Float16* As, int rowInGroup,
                                           int halfsel, int kchunk, Frag& af) {
    const int base = rowInGroup * K_PAD + kchunk * 32 + halfsel * 8;
#pragma unroll
    for (int v = 0; v < 8; ++v) {
        const int k = ((v < 4) ? 0 : 16) + ((v & 3) * 2);
        af.u[v] = *(const unsigned int*)&As[base + k];
    }
}

// stage 48 rows x 192 halves of [self | nbr | nbr_fea] into LDS as f16
__device__ __forceinline__ void stage_A(_Float16* As, int grp,
                                        const float* __restrict__ atomIn,
                                        const float* __restrict__ nbr_fea,
                                        const int* __restrict__ nbr_idx) {
    const int n0 = grp * G_ATOMS;
    for (int idx = threadIdx.x; idx < G_ROWS * K_PAD; idx += 256) {
        const int row = idx / K_PAD, k = idx - row * K_PAD;
        const int a = row / M_NBR, m = row - a * M_NBR;
        const int n = n0 + a;
        float val = 0.f;
        if (k < F_DIM) {
            val = atomIn[(size_t)n * F_DIM + k];
        } else if (k < 2 * F_DIM) {
            const int j = nbr_idx[n * M_NBR + m];
            val = atomIn[(size_t)j * F_DIM + (k - F_DIM)];
        } else if (k < K_DIM) {
            val = nbr_fea[(size_t)(n * M_NBR + m) * NBR_F + (k - 2 * F_DIM)];
        }
        As[idx] = (_Float16)val;
    }
}

// ------------------- conv pass 1: BN1 column stats only --------------------
__global__ __launch_bounds__(256)
void conv_pass1(const float* __restrict__ atomIn, const float* __restrict__ nbr_fea,
                const int* __restrict__ nbr_idx, const unsigned int* __restrict__ wpack,
                const float* __restrict__ bias,
                float* __restrict__ s1sum, float* __restrict__ s1q) {
    __shared__ _Float16 As[G_ROWS * K_PAD];
    const int tid = threadIdx.x;
    const int wave = tid >> 5, lane = tid & 31;
    const int cit = lane & 15, halfsel = lane >> 4;
    const int col = wave * 16 + cit;
    const float biasc = bias[col];

    Frag bf[6];
#pragma unroll
    for (int c = 0; c < 6; ++c)
#pragma unroll
        for (int v = 0; v < 8; ++v)
            bf[c].u[v] = wpack[((c * 8 + wave) * 32 + lane) * 8 + v];

    float sAcc = 0.f, qAcc = 0.f;

    for (int grp = blockIdx.x; grp < N_GROUPS; grp += gridDim.x) {
        stage_A(As, grp, atomIn, nbr_fea, nbr_idx);
        __syncthreads();
#pragma unroll
        for (int f = 0; f < 3; ++f) {
            v8f acc = {};
#pragma unroll
            for (int c = 0; c < 6; ++c) {
                Frag af;
                load_afrag(As, f * 16 + cit, halfsel, c, af);
                acc = __builtin_amdgcn_wmma_f32_16x16x32_f16(
                    false, af.v, false, bf[c].v, (short)0, acc, false, false);
            }
#pragma unroll
            for (int v = 0; v < 8; ++v) {
                const float g = acc[v] + biasc;
                sAcc += g; qAcc += g * g;
            }
        }
        __syncthreads();
    }
    // lanes l and l+16 hold the same output column
    sAcc += __shfl_xor(sAcc, 16);
    qAcc += __shfl_xor(qAcc, 16);
    if (halfsel == 0) {
        atomicAdd(&s1sum[col], sAcc);
        atomicAdd(&s1q[col],  qAcc);
    }
}

// ------ conv pass 2: recompute, BN1, sigmoid*softplus, sum_m, BN2 stats ----
__global__ __launch_bounds__(256)
void conv_pass2(const float* __restrict__ atomIn, const float* __restrict__ nbr_fea,
                const int* __restrict__ nbr_idx, const unsigned int* __restrict__ wpack,
                const float* __restrict__ bias,
                const float* __restrict__ g1, const float* __restrict__ b1,
                const float* __restrict__ mean1, const float* __restrict__ inv1,
                float* __restrict__ summed,
                float* __restrict__ s2sum, float* __restrict__ s2q) {
    __shared__ _Float16 As[G_ROWS * K_PAD];
    __shared__ float gbuf[G_ROWS * GSTRIDE];
    __shared__ float red[256];
    const int tid = threadIdx.x;
    const int wave = tid >> 5, lane = tid & 31;
    const int cit = lane & 15, halfsel = lane >> 4;
    const int col = wave * 16 + cit;
    const float biasc = bias[col];
    const float m1 = mean1[col], iv1 = inv1[col];
    const float gg = g1[col], bb = b1[col];

    Frag bf[6];
#pragma unroll
    for (int c = 0; c < 6; ++c)
#pragma unroll
        for (int v = 0; v < 8; ++v)
            bf[c].u[v] = wpack[((c * 8 + wave) * 32 + lane) * 8 + v];

    const int fcol = tid & 63;          // feature for summed
    const int arow = tid >> 6;          // atom-in-group (0..3)
    float s2Acc = 0.f, q2Acc = 0.f;

    for (int grp = blockIdx.x; grp < N_GROUPS; grp += gridDim.x) {
        stage_A(As, grp, atomIn, nbr_fea, nbr_idx);
        __syncthreads();
#pragma unroll
        for (int f = 0; f < 3; ++f) {
            v8f acc = {};
#pragma unroll
            for (int c = 0; c < 6; ++c) {
                Frag af;
                load_afrag(As, f * 16 + cit, halfsel, c, af);
                acc = __builtin_amdgcn_wmma_f32_16x16x32_f16(
                    false, af.v, false, bf[c].v, (short)0, acc, false, false);
            }
            // C/D layout: lane col = lane&15, rows = (lane>>4)*8 + v
#pragma unroll
            for (int v = 0; v < 8; ++v) {
                const int row = f * 16 + halfsel * 8 + v;
                const float g = acc[v] + biasc;
                gbuf[row * GSTRIDE + col] = gg * (g - m1) * iv1 + bb;
            }
        }
        __syncthreads();
        // sigmoid(filter)*softplus(core), summed over the 12 neighbors
        {
            const int n = grp * G_ATOMS + arow;
            float accv = 0.f;
#pragma unroll
            for (int m = 0; m < M_NBR; ++m) {
                const int row = arow * M_NBR + m;
                const float flt = gbuf[row * GSTRIDE + fcol];
                const float cor = gbuf[row * GSTRIDE + 64 + fcol];
                accv += sigmoidf(flt) * softplusf(cor);
            }
            summed[(size_t)n * F_DIM + fcol] = accv;
            s2Acc += accv; q2Acc += accv * accv;
        }
        __syncthreads();
    }
    // block-reduce BN2 stats (4 threads per feature) then one atomic each
    red[tid] = s2Acc; __syncthreads();
    if (tid < 64) atomicAdd(&s2sum[tid], red[tid] + red[tid + 64] + red[tid + 128] + red[tid + 192]);
    __syncthreads();
    red[tid] = q2Acc; __syncthreads();
    if (tid < 64) atomicAdd(&s2q[tid],  red[tid] + red[tid + 64] + red[tid + 128] + red[tid + 192]);
}

// ---------------- atom_new = softplus(atom + BN2(summed)) ------------------
__global__ void bn2_softplus(const float* __restrict__ atomIn, const float* __restrict__ summed,
                             const float* __restrict__ g2, const float* __restrict__ b2,
                             const float* __restrict__ mean2, const float* __restrict__ inv2,
                             float* __restrict__ atomOut) {
    int idx = blockIdx.x * blockDim.x + threadIdx.x;
    if (idx >= N_ATOMS * F_DIM) return;
    int f = idx & 63;
    float v = g2[f] * (summed[idx] - mean2[f]) * inv2[f] + b2[f];
    atomOut[idx] = softplusf(atomIn[idx] + v);
}

// ------------------------------ pooling ------------------------------------
__global__ void pool_kernel(const float* __restrict__ atomF, const int* __restrict__ batch,
                            float* __restrict__ pooled, float* __restrict__ cnts) {
    int idx = blockIdx.x * blockDim.x + threadIdx.x;
    if (idx >= N_ATOMS * F_DIM) return;
    int n = idx >> 6, f = idx & 63;
    int b = batch[n];
    atomicAdd(&pooled[(size_t)b * F_DIM + f], atomF[idx]);
    if (f == 0) atomicAdd(&cnts[b], 1.f);
}

__global__ void pool_avg(const float* __restrict__ pooled, const float* __restrict__ cnts,
                         float* __restrict__ xc) {
    int idx = blockIdx.x * blockDim.x + threadIdx.x;
    if (idx >= B_CRYS * F_DIM) return;
    int b = idx >> 6;
    xc[idx] = pooled[idx] / fmaxf(cnts[b], 1.f);
}

// ----------------------- small dense stack (B=2000) ------------------------
__global__ void lin_kernel(const float* __restrict__ X, const float* __restrict__ W,
                           const float* __restrict__ bias, float* __restrict__ Y,
                           int rows, int K, int Dout) {
    int idx = blockIdx.x * blockDim.x + threadIdx.x;
    if (idx >= rows * Dout) return;
    int r = idx / Dout, c = idx - r * Dout;
    float a = bias[c];
    const float* xr = X + (size_t)r * K;
#pragma unroll 4
    for (int k = 0; k < K; ++k) a += xr[k] * W[(size_t)k * Dout + c];
    Y[idx] = a;
}

__global__ __launch_bounds__(256)
void colstats(const float* __restrict__ Y, int rows, int Dout,
              float* __restrict__ mean, float* __restrict__ inv) {
    __shared__ float ss[256], qq[256];
    int c = blockIdx.x;
    float s = 0.f, q = 0.f;
    for (int r = threadIdx.x; r < rows; r += 256) {
        float v = Y[(size_t)r * Dout + c];
        s += v; q += v * v;
    }
    ss[threadIdx.x] = s; qq[threadIdx.x] = q; __syncthreads();
    for (int o = 128; o > 0; o >>= 1) {
        if (threadIdx.x < o) { ss[threadIdx.x] += ss[threadIdx.x + o]; qq[threadIdx.x] += qq[threadIdx.x + o]; }
        __syncthreads();
    }
    if (threadIdx.x == 0) {
        float m = ss[0] / rows;
        float v = qq[0] / rows - m * m;
        mean[c] = m; inv[c] = rsqrtf(v + EPSV);
    }
}

__global__ void bnrelu(const float* __restrict__ Y, const float* __restrict__ g,
                       const float* __restrict__ b, const float* __restrict__ mean,
                       const float* __restrict__ inv, float* __restrict__ X,
                       int n, int Dout) {
    int idx = blockIdx.x * blockDim.x + threadIdx.x;
    if (idx >= n) return;
    int c = idx % Dout;
    float y = g[c] * (Y[idx] - mean[c]) * inv[c] + b[c];
    X[idx] = fmaxf(y, 0.f);
}

__global__ void regression(const float* __restrict__ X, const float* __restrict__ wr,
                           const float* __restrict__ br, float* __restrict__ out) {
    int b = blockIdx.x * blockDim.x + threadIdx.x;
    if (b >= B_CRYS) return;
    float a = br[0];
    const float* xr = X + (size_t)b * D_DIM;
#pragma unroll 4
    for (int k = 0; k < D_DIM; ++k) a += xr[k] * wr[k];
    out[b] = a;
}

// ---------------------------------------------------------------------------
extern "C" void kernel_launch(void* const* d_in, const int* in_sizes, int n_in,
                              void* d_out, int out_size, void* d_ws, size_t ws_size,
                              hipStream_t stream) {
    const float* x        = (const float*)d_in[0];
    const float* nbr_fea  = (const float*)d_in[1];
    const int*   nbr_idx  = (const int*)d_in[2];
    const int*   batch    = (const int*)d_in[3];
    const float* w1       = (const float*)d_in[4];
    const float* b1       = (const float*)d_in[5];
    const float* conv_w   = (const float*)d_in[6];
    const float* conv_b   = (const float*)d_in[7];
    const float* bn1_g    = (const float*)d_in[8];
    const float* bn1_b    = (const float*)d_in[9];
    const float* bn2_g    = (const float*)d_in[10];
    const float* bn2_b    = (const float*)d_in[11];
    const float* w2       = (const float*)d_in[12];
    const float* b2       = (const float*)d_in[13];
    const float* bnin_g   = (const float*)d_in[14];
    const float* bnin_b   = (const float*)d_in[15];
    const float* wc1      = (const float*)d_in[16];
    const float* bc1      = (const float*)d_in[17];
    const float* bnc1_g   = (const float*)d_in[18];
    const float* bnc1_b   = (const float*)d_in[19];
    const float* wc2      = (const float*)d_in[20];
    const float* bc2      = (const float*)d_in[21];
    const float* bnc2_g   = (const float*)d_in[22];
    const float* bnc2_b   = (const float*)d_in[23];
    const float* wr       = (const float*)d_in[24];
    const float* br       = (const float*)d_in[25];
    float* out = (float*)d_out;

    // workspace carve-up
    char* ws = (char*)d_ws;
    size_t off = 0;
    auto carve = [&](size_t bytes) {
        void* p = ws + off;
        off = (off + bytes + 255) & ~size_t(255);
        return p;
    };
    float* atomA  = (float*)carve((size_t)N_ATOMS * F_DIM * 4);
    float* atomB  = (float*)carve((size_t)N_ATOMS * F_DIM * 4);
    float* summed = (float*)carve((size_t)N_ATOMS * F_DIM * 4);
    unsigned int* wpack = (unsigned int*)carve((size_t)3 * WPACK_PER_LAYER * 4);
    float* s1sum = (float*)carve(128 * 4);
    float* s1q   = (float*)carve(128 * 4);
    float* mean1 = (float*)carve(128 * 4);
    float* inv1  = (float*)carve(128 * 4);
    float* s2sum = (float*)carve(64 * 4);
    float* s2q   = (float*)carve(64 * 4);
    float* mean2 = (float*)carve(64 * 4);
    float* inv2  = (float*)carve(64 * 4);
    float* pooled = (float*)carve((size_t)B_CRYS * F_DIM * 4);
    float* cnts   = (float*)carve((size_t)B_CRYS * 4);
    float* xc     = (float*)carve((size_t)B_CRYS * F_DIM * 4);
    float* Ybuf   = (float*)carve((size_t)B_CRYS * D_DIM * 4);
    float* X1     = (float*)carve((size_t)B_CRYS * D_DIM * 4);
    float* X2     = (float*)carve((size_t)B_CRYS * D_DIM * 4);
    float* X3     = (float*)carve((size_t)B_CRYS * D_DIM * 4);
    float* dmean  = (float*)carve(128 * 4);
    float* dinv   = (float*)carve(128 * 4);
    (void)ws_size; (void)n_in; (void)in_sizes; (void)out_size;

    const int NF = N_ATOMS * F_DIM;

    // pack weights + embed
    pack_weights<<<(3 * WPACK_PER_LAYER + 255) / 256, 256, 0, stream>>>(conv_w, wpack);
    embed_kernel<<<(NF + 255) / 256, 256, 0, stream>>>(x, w1, b1, atomA);

    float* cur = atomA;
    float* nxt = atomB;
    for (int i = 0; i < 3; ++i) {
        const unsigned int* wp = wpack + (size_t)i * WPACK_PER_LAYER;
        const float* bias = conv_b + i * D_DIM;
        zero_f32<<<1, 256, 0, stream>>>(s1sum, 128);
        zero_f32<<<1, 256, 0, stream>>>(s1q, 128);
        conv_pass1<<<1024, 256, 0, stream>>>(cur, nbr_fea, nbr_idx, wp, bias, s1sum, s1q);
        finalize_stats<<<1, 128, 0, stream>>>(s1sum, s1q, (float)(N_ATOMS * M_NBR), 128, mean1, inv1);
        zero_f32<<<1, 128, 0, stream>>>(s2sum, 64);
        zero_f32<<<1, 128, 0, stream>>>(s2q, 64);
        conv_pass2<<<1024, 256, 0, stream>>>(cur, nbr_fea, nbr_idx, wp, bias,
                                             bn1_g + i * D_DIM, bn1_b + i * D_DIM,
                                             mean1, inv1, summed, s2sum, s2q);
        finalize_stats<<<1, 64, 0, stream>>>(s2sum, s2q, (float)N_ATOMS, 64, mean2, inv2);
        bn2_softplus<<<(NF + 255) / 256, 256, 0, stream>>>(cur, summed,
                                                           bn2_g + i * F_DIM, bn2_b + i * F_DIM,
                                                           mean2, inv2, nxt);
        float* t = cur; cur = nxt; nxt = t;
    }
    // cur now holds final atom features

    zero_f32<<<(B_CRYS * F_DIM + 255) / 256, 256, 0, stream>>>(pooled, B_CRYS * F_DIM);
    zero_f32<<<(B_CRYS + 255) / 256, 256, 0, stream>>>(cnts, B_CRYS);
    pool_kernel<<<(NF + 255) / 256, 256, 0, stream>>>(cur, batch, pooled, cnts);
    pool_avg<<<(B_CRYS * F_DIM + 255) / 256, 256, 0, stream>>>(pooled, cnts, xc);

    const int BD = B_CRYS * D_DIM;
    // xc @ w2 + b2 -> BN -> relu
    lin_kernel<<<(BD + 255) / 256, 256, 0, stream>>>(xc, w2, b2, Ybuf, B_CRYS, F_DIM, D_DIM);
    colstats<<<D_DIM, 256, 0, stream>>>(Ybuf, B_CRYS, D_DIM, dmean, dinv);
    bnrelu<<<(BD + 255) / 256, 256, 0, stream>>>(Ybuf, bnin_g, bnin_b, dmean, dinv, X1, BD, D_DIM);
    // crystal_lin1
    lin_kernel<<<(BD + 255) / 256, 256, 0, stream>>>(X1, wc1, bc1, Ybuf, B_CRYS, D_DIM, D_DIM);
    colstats<<<D_DIM, 256, 0, stream>>>(Ybuf, B_CRYS, D_DIM, dmean, dinv);
    bnrelu<<<(BD + 255) / 256, 256, 0, stream>>>(Ybuf, bnc1_g, bnc1_b, dmean, dinv, X2, BD, D_DIM);
    // crystal_lin2
    lin_kernel<<<(BD + 255) / 256, 256, 0, stream>>>(X2, wc2, bc2, Ybuf, B_CRYS, D_DIM, D_DIM);
    colstats<<<D_DIM, 256, 0, stream>>>(Ybuf, B_CRYS, D_DIM, dmean, dinv);
    bnrelu<<<(BD + 255) / 256, 256, 0, stream>>>(Ybuf, bnc2_g, bnc2_b, dmean, dinv, X3, BD, D_DIM);
    // regression
    regression<<<(B_CRYS + 255) / 256, 256, 0, stream>>>(X3, wr, br, out);
}
